// SimpleGraphUNet_867583393849
// MI455X (gfx1250) — matrix operand order repaired and checked by reading, not compiled, not run on hardware
//
#include <hip/hip_runtime.h>

typedef __attribute__((ext_vector_type(16))) __bf16 v16bf;
typedef __attribute__((ext_vector_type(8)))  __bf16 v8bf;
typedef __attribute__((ext_vector_type(8)))  float  v8f;

#define NN 100000
#define NE 1600000
#define NG 128
#define BN_EPS 1e-5f

static __device__ __forceinline__ int gtid() { return blockIdx.x * blockDim.x + threadIdx.x; }

__global__ void k_zero(float* __restrict__ p, int n) {
  int i = gtid(); if (i < n) p[i] = 0.0f;
}

__global__ void k_cvt_bf16(const float* __restrict__ in, __bf16* __restrict__ out, int n) {
  int i = gtid(); if (i < n) out[i] = (__bf16)in[i];
}

// actbf = bf16(h + agg)   (GIN input:  x + sum_neighbors)
__global__ void k_add_cvt_bf16(const float* __restrict__ a, const float* __restrict__ b,
                               __bf16* __restrict__ out, int n) {
  int i = gtid(); if (i < n) out[i] = (__bf16)(a[i] + b[i]);
}

// agg[dst] += h[src], 4 feats per thread (coalesced row reads, f32 atomics to L2)
__global__ void k_scatter(const float* __restrict__ h, const int* __restrict__ src,
                          const int* __restrict__ dst, float* __restrict__ agg,
                          int dim4, int dim, int total) {
  int t = gtid(); if (t >= total) return;
  int e = t / dim4;
  int q = (t - e * dim4) * 4;
  int s = src[e], d = dst[e];
  const float4 v = *(const float4*)(h + (size_t)s * dim + q);
  float* o = agg + (size_t)d * dim + q;
  atomicAdd(o + 0, v.x); atomicAdd(o + 1, v.y);
  atomicAdd(o + 2, v.z); atomicAdd(o + 3, v.w);
}

// C[N x dout] = A[N x din](bf16) * W[dout x din]^T(bf16) + bias, f32 accum via WMMA.
// One 16x64 output strip per wave (4 N-tiles share each A fragment -> 4x A reuse).
__global__ __launch_bounds__(256) void k_gemm(const __bf16* __restrict__ A,
                                              const __bf16* __restrict__ W,
                                              const float* __restrict__ bias,
                                              float* __restrict__ C,
                                              int din, int dout, int tilesn4, int total_strips) {
  int wid = (blockIdx.x * blockDim.x + threadIdx.x) >> 5;   // wave-uniform
  if (wid >= total_strips) return;                          // whole-wave exit; EXEC stays all-1s
  int lane = threadIdx.x & 31;
  int half = lane >> 4;          // 0: K low block, 1: K high block (contiguous k order, same in A and B)
  int l15  = lane & 15;
  int tm = wid / tilesn4, tn4 = wid - tm * tilesn4;

  // A frag: lane<16 -> row, K {k..k+15}; lane>=16 -> K {k+16..k+31} (contiguous, matches B)
  const __bf16* ap = A + (size_t)(tm * 16 + l15) * din + half * 16;
  // B frags: lane n -> output column (= row of W); K contiguous (16 elems per half)
  int col0 = tn4 * 64 + l15;
  const __bf16* bp = W + (size_t)col0 * din + half * 16;
  size_t bstride = (size_t)16 * din;   // 16 output columns apart

  v8f a0 = {0.f,0.f,0.f,0.f,0.f,0.f,0.f,0.f};
  v8f a1 = a0, a2 = a0, a3 = a0;
  for (int k = 0; k < din; k += 32) {
    v8bf alo = *(const v8bf*)(ap + k);
    v8bf ahi = *(const v8bf*)(ap + k + 8);
    v16bf av = __builtin_shufflevector(alo, ahi, 0,1,2,3,4,5,6,7,8,9,10,11,12,13,14,15);
    v16bf b0 = *(const v16bf*)(bp + k);
    v16bf b1 = *(const v16bf*)(bp + bstride + k);
    v16bf b2 = *(const v16bf*)(bp + 2 * bstride + k);
    v16bf b3 = *(const v16bf*)(bp + 3 * bstride + k);
    a0 = __builtin_amdgcn_wmma_f32_16x16x32_bf16(false, av, false, b0, (short)0, a0, false, false);
    a1 = __builtin_amdgcn_wmma_f32_16x16x32_bf16(false, av, false, b1, (short)0, a1, false, false);
    a2 = __builtin_amdgcn_wmma_f32_16x16x32_bf16(false, av, false, b2, (short)0, a2, false, false);
    a3 = __builtin_amdgcn_wmma_f32_16x16x32_bf16(false, av, false, b3, (short)0, a3, false, false);
  }
  // D layout: lane n<16 -> rows +0..7 ; lane n>=16 -> rows +8..15 ; col = strip base + j*16 + l15
  float* cbase = C + (size_t)(tm * 16 + half * 8) * dout + col0;
  float bb0 = bias[col0], bb1 = bias[col0 + 16], bb2 = bias[col0 + 32], bb3 = bias[col0 + 48];
#pragma unroll
  for (int r = 0; r < 8; r++) {
    float* cp = cbase + (size_t)r * dout;
    cp[0]  = a0[r] + bb0;
    cp[16] = a1[r] + bb1;
    cp[32] = a2[r] + bb2;
    cp[48] = a3[r] + bb3;
  }
}

// Per-column sum / sumsq (dim divides 256 -> fully coalesced row-band reads)
__global__ void k_colstats(const float* __restrict__ Y, float* __restrict__ sum,
                           float* __restrict__ sumsq, int dim, int n) {
  int col  = threadIdx.x % dim;
  int rsub = threadIdx.x / dim;
  int rpb  = blockDim.x / dim;
  float s = 0.f, q = 0.f;
  for (int r = blockIdx.x * rpb + rsub; r < n; r += gridDim.x * rpb) {
    float v = Y[(size_t)r * dim + col];
    s += v; q += v * v;
  }
  atomicAdd(&sum[col], s);
  atomicAdd(&sumsq[col], q);
}

__global__ void k_bnfin(const float* __restrict__ sum, const float* __restrict__ sumsq,
                        const float* __restrict__ g, const float* __restrict__ be,
                        float* __restrict__ scale, float* __restrict__ shift,
                        int dim, float inv_n) {
  int c = gtid(); if (c >= dim) return;
  float m = sum[c] * inv_n;
  float v = sumsq[c] * inv_n - m * m;
  float a = g[c] * rsqrtf(v + BN_EPS);
  scale[c] = a;
  shift[c] = be[c] - m * a;
}

// out = relu(Y*scale + shift) [+ skip]; writes bf16 and/or f32 (uniform null checks)
__global__ void k_bnapply(const float* __restrict__ Y, const float* __restrict__ scale,
                          const float* __restrict__ shift, const float* __restrict__ skip,
                          __bf16* __restrict__ obf, float* __restrict__ of,
                          int dim, int total) {
  int i = gtid(); if (i >= total) return;
  int c = i % dim;
  float v = fmaxf(fmaf(Y[i], scale[c], shift[c]), 0.0f);
  if (skip) v += skip[i];
  if (of)  of[i] = v;
  if (obf) obf[i] = (__bf16)v;
}

__global__ void k_counts(const int* __restrict__ batch, float* __restrict__ cnt) {
  int n = gtid(); if (n < NN) atomicAdd(&cnt[batch[n]], 1.0f);
}

// x_d1 = X(N x 64) @ Wd1^T + b, scattered straight into per-graph sums
__global__ void k_d1pool(const float* __restrict__ X, const float* __restrict__ W,
                         const float* __restrict__ b, const int* __restrict__ batch,
                         float* __restrict__ psum) {
  int t = gtid();
  int n = t / 10, c = t - n * 10;
  if (n >= NN) return;
  const float* xr = X + (size_t)n * 64;
  const float* wr = W + c * 64;
  float acc = b[c];
#pragma unroll
  for (int k = 0; k < 64; k++) acc = fmaf(xr[k], wr[k], acc);
  atomicAdd(&psum[batch[n] * 10 + c], acc);
}

__global__ void k_poolfin(const float* __restrict__ psum, const float* __restrict__ cnt,
                          float* __restrict__ out) {
  int i = gtid(); if (i >= NG * 10) return;
  out[i] = psum[i] / fmaxf(cnt[i / 10], 1.0f);
}

extern "C" void kernel_launch(void* const* d_in, const int* in_sizes, int n_in,
                              void* d_out, int out_size, void* d_ws, size_t ws_size,
                              hipStream_t stream) {
  const float* x   = (const float*)d_in[0];
  const int* ei    = (const int*)d_in[1];
  const int* src   = ei;
  const int* dst   = ei + NE;
  const int* batch = (const int*)d_in[2];
  auto F = [&](int i) { return (const float*)d_in[i]; };

  // ---- workspace carve-up (~282 MB) ----
  char* ws = (char*)d_ws;
  size_t off = 0;
  auto alloc = [&](size_t bytes) -> char* {
    char* p = ws + off; off += (bytes + 255) & ~(size_t)255; return p;
  };
  float*  agg   = (float*)alloc((size_t)NN * 128 * 4);  // neighbor sums; reused as final f32
  float*  bufA  = (float*)alloc((size_t)NN * 256 * 4);  // pre-BN linear outputs
  float*  x1    = (float*)alloc((size_t)NN * 64 * 4);
  float*  x2    = (float*)alloc((size_t)NN * 128 * 4);
  __bf16* actbf = (__bf16*)alloc((size_t)NN * 256 * 2); // bf16 GEMM input staging
  __bf16* wbf   = (__bf16*)alloc((size_t)196608 * 2);   // all weights in bf16
  float*  ssum  = (float*)alloc(256 * 4);
  float*  ssq   = (float*)alloc(256 * 4);
  float*  sscl  = (float*)alloc(256 * 4);
  float*  ssft  = (float*)alloc(256 * 4);
  float*  psum  = (float*)alloc(NG * 10 * 4);
  float*  pcnt  = (float*)alloc(NG * 4);

  auto zero = [&](float* p, int n) { k_zero<<<(n + 255) / 256, 256, 0, stream>>>(p, n); };

  // params flatten order: x, edge_index, batch, then per MLP {W1,b1,g1,be1,W2,b2,g2,be2}
  // for c1,c2,c3,d3,d2 (base 3+8m), then d1 {W=43,b=44}
  struct WM { int idx, offE, sz; };
  const WM wm[10] = {
    { 3,      0,  64 * 128}, { 7,   8192,  64 * 64 },
    {11,  12288, 128 * 64 }, {15,  20480, 128 * 128},
    {19,  36864, 256 * 128}, {23,  69632, 256 * 256},
    {27, 135168, 128 * 256}, {31, 167936, 128 * 128},
    {35, 184320,  64 * 128}, {39, 192512,  64 * 64 },
  };
  for (int i = 0; i < 10; i++)
    k_cvt_bf16<<<(wm[i].sz + 255) / 256, 256, 0, stream>>>(F(wm[i].idx), wbf + wm[i].offE, wm[i].sz);

  auto gemm = [&](int wslot, const float* bias, float* C, int din, int dout) {
    int tilesn4 = dout / 64;              // 16x64 strip per wave; all douts divide by 64
    int total   = (NN / 16) * tilesn4;    // NN % 16 == 0
    k_gemm<<<(total + 7) / 8, 256, 0, stream>>>(actbf, wbf + wslot, bias, C,
                                                din, dout, tilesn4, total);
  };
  auto bnstage = [&](float* Y, const float* g, const float* be, int dim,
                     const float* skip, __bf16* obf, float* of) {
    zero(ssum, dim); zero(ssq, dim);
    k_colstats<<<1024, 256, 0, stream>>>(Y, ssum, ssq, dim, NN);
    k_bnfin<<<1, 256, 0, stream>>>(ssum, ssq, g, be, sscl, ssft, dim, 1.0f / NN);
    int total = NN * dim;
    k_bnapply<<<(total + 255) / 256, 256, 0, stream>>>(Y, sscl, ssft, skip, obf, of, dim, total);
  };
  auto ginagg = [&](const float* h, int dim) {
    zero(agg, NN * dim);
    int dim4 = dim / 4, total = NE * dim4;
    k_scatter<<<(total + 255) / 256, 256, 0, stream>>>(h, src, dst, agg, dim4, dim, total);
    int t2 = NN * dim;
    k_add_cvt_bf16<<<(t2 + 255) / 256, 256, 0, stream>>>(h, agg, actbf, t2);
  };

  // ---- encoder: GIN convs ----
  ginagg(x, 128);                                               // c1
  gemm(     0, F(4),  bufA, 128,  64);
  bnstage(bufA, F(5),  F(6),  64, nullptr, actbf, nullptr);
  gemm(  8192, F(8),  bufA,  64,  64);
  bnstage(bufA, F(9),  F(10), 64, nullptr, nullptr, x1);        // x1 (f32 kept for skip)

  ginagg(x1, 64);                                               // c2
  gemm( 12288, F(12), bufA,  64, 128);
  bnstage(bufA, F(13), F(14), 128, nullptr, actbf, nullptr);
  gemm( 20480, F(16), bufA, 128, 128);
  bnstage(bufA, F(17), F(18), 128, nullptr, nullptr, x2);       // x2 (f32 kept for skip)

  ginagg(x2, 128);                                              // c3
  gemm( 36864, F(20), bufA, 128, 256);
  bnstage(bufA, F(21), F(22), 256, nullptr, actbf, nullptr);
  gemm( 69632, F(24), bufA, 256, 256);
  bnstage(bufA, F(25), F(26), 256, nullptr, actbf, nullptr);    // x3 only needed as bf16

  // ---- decoder ----
  gemm(135168, F(28), bufA, 256, 128);                          // d3
  bnstage(bufA, F(29), F(30), 128, nullptr, actbf, nullptr);
  gemm(167936, F(32), bufA, 128, 128);
  bnstage(bufA, F(33), F(34), 128, x2, actbf, nullptr);         // (x_d3 + x2) -> bf16

  gemm(184320, F(36), bufA, 128, 64);                           // d2
  bnstage(bufA, F(37), F(38), 64, nullptr, actbf, nullptr);
  gemm(192512, F(40), bufA, 64, 64);
  bnstage(bufA, F(41), F(42), 64, x1, nullptr, agg);            // (x_d2 + x1) f32 (agg reused)

  // ---- d1 linear fused with global mean pool ----
  zero(psum, NG * 10); zero(pcnt, NG);
  k_counts<<<(NN + 255) / 256, 256, 0, stream>>>(batch, pcnt);
  k_d1pool<<<(NN * 10 + 255) / 256, 256, 0, stream>>>(agg, F(43), F(44), batch, psum);
  k_poolfin<<<(NG * 10 + 255) / 256, 256, 0, stream>>>(psum, pcnt, (float*)d_out);
}
